// DAGTaskEncoder_26431228740058
// MI455X (gfx1250) — matrix operand order repaired and checked by reading, not compiled, not run on hardware
//
#include <hip/hip_runtime.h>

#define NEG_SLOPE 0.2f

typedef __attribute__((ext_vector_type(16))) __bf16 v16bf;
typedef __attribute__((ext_vector_type(8)))  float  v8f;

// ---------- monotone float<->uint encoding for atomicMax on floats ----------
__device__ __forceinline__ unsigned fenc(float f) {
  unsigned u = __float_as_uint(f);
  return (u & 0x80000000u) ? ~u : (u | 0x80000000u);
}
__device__ __forceinline__ float fdec(unsigned u) {
  return (u & 0x80000000u) ? __uint_as_float(u & 0x7FFFFFFFu) : __uint_as_float(~u);
}
#define ENC_NEG_INF 0x007FFFFFu   // fenc(-inf)

// ---------- init ----------
__global__ void init_u32(unsigned* p, unsigned v, int n) {
  int i = blockIdx.x * blockDim.x + threadIdx.x;
  if (i < n) p[i] = v;
}

// ---------- WMMA GEMM: Y[nrows x 128] = X[nrows x 128] @ W[128 x 128] ----------
// Block = 256 threads = 8 waves. Block handles one 16-row band; wave w handles
// the 16-column tile [16w, 16w+16). K swept 0..128 in steps of 32 (bf16 WMMA).
__global__ void gemm128(const float* __restrict__ X, const float* __restrict__ W,
                        float* __restrict__ Y, int nrows) {
  const int wave = threadIdx.x >> 5;
  const int lane = threadIdx.x & 31;
  const bool hi  = lane >= 16;
  const int  l   = lane & 15;

  const int m0 = blockIdx.x * 16;
  const int n0 = wave * 16;

  int arow = m0 + l;
  if (arow >= nrows) arow = nrows - 1;          // clamp (no divergence around WMMA)
  const float* xrow = X + (size_t)arow * 128;
  const int ncol = n0 + l;

  v8f acc = {};
  #pragma unroll
  for (int k0 = 0; k0 < 128; k0 += 32) {
    // A fragment 16x32 bf16 (ISA 7.12.2): lanes 0-15 row=l hold K=k0+0..7 (v0..3)
    // and K=k0+16..23 (v4..7); lanes 16-31 hold K offsets +8 / +24.
    v16bf a;
    const int ka = k0 + (hi ? 8 : 0);
    #pragma unroll
    for (int j = 0; j < 4; ++j) {
      float2 p = *(const float2*)(xrow + ka + 2 * j);
      a[2 * j]     = (__bf16)p.x;
      a[2 * j + 1] = (__bf16)p.y;
      float2 q = *(const float2*)(xrow + ka + 16 + 2 * j);
      a[8 + 2 * j]     = (__bf16)q.x;
      a[8 + 2 * j + 1] = (__bf16)q.y;
    }
    // B fragment 32x16 bf16 (same K order as A): lanes 0-15 (col=l) hold K=k0+0..7, k0+16..23;
    // lanes 16-31 hold K offsets +8 / +24.
    v16bf b;
    const int kb = k0 + (hi ? 8 : 0);
    #pragma unroll
    for (int j = 0; j < 8; ++j) {
      const int kk = kb + (j < 4 ? 2 * j : 8 + 2 * j);
      b[2 * j]     = (__bf16)W[(size_t)(kk) * 128 + ncol];
      b[2 * j + 1] = (__bf16)W[(size_t)(kk + 1) * 128 + ncol];
    }
    acc = __builtin_amdgcn_wmma_f32_16x16x32_bf16(false, a, false, b,
                                                  (short)0, acc, false, false);
  }
  // C/D layout: VGPR r -> (M=r, N=lane) lanes 0-15 ; (M=r+8, N=lane-16) lanes 16-31
  const int mstore = m0 + (hi ? 8 : 0);
  #pragma unroll
  for (int r = 0; r < 8; ++r) {
    int row = mstore + r;
    if (row < nrows) Y[(size_t)row * 128 + ncol] = acc[r];
  }
}

// ---------- pass 1: edge scores + segment max (wave per edge) ----------
template <int HEADS>
__global__ void edge_score_max(const float* __restrict__ xl, const float* __restrict__ xr,
                               const int* __restrict__ srcE, const int* __restrict__ dstE,
                               const float* __restrict__ att,
                               unsigned* __restrict__ emax, float* __restrict__ escore,
                               int E_orig, int E_tot) {
  const int widx = blockIdx.x * (blockDim.x >> 5) + (threadIdx.x >> 5);
  if (widx >= E_tot) return;
  const int lane = threadIdx.x & 31;
  int s, d;
  if (widx < E_orig) { s = srcE[widx]; d = dstE[widx]; } else { s = d = widx - E_orig; }

  const float4 lv = *(const float4*)(xl + (size_t)s * 128 + lane * 4);
  const float4 rv = *(const float4*)(xr + (size_t)d * 128 + lane * 4);
  const float4 av = *(const float4*)(att + lane * 4);

  float m0 = lv.x + rv.x, m1 = lv.y + rv.y, m2 = lv.z + rv.z, m3 = lv.w + rv.w;
  m0 = m0 > 0.f ? m0 : m0 * NEG_SLOPE;
  m1 = m1 > 0.f ? m1 : m1 * NEG_SLOPE;
  m2 = m2 > 0.f ? m2 : m2 * NEG_SLOPE;
  m3 = m3 > 0.f ? m3 : m3 * NEG_SLOPE;
  float part = av.x * m0 + av.y * m1 + av.z * m2 + av.w * m3;

  // reduce within each 16-lane half (head) — or whole wave for HEADS==1
  part += __shfl_xor(part, 1);
  part += __shfl_xor(part, 2);
  part += __shfl_xor(part, 4);
  part += __shfl_xor(part, 8);
  if (HEADS == 1) part += __shfl_xor(part, 16);

  if (HEADS == 2) {
    if ((lane & 15) == 0) {
      int h = lane >> 4;
      escore[(size_t)widx * 2 + h] = part;
      atomicMax(&emax[(size_t)d * 2 + h], fenc(part));
    }
  } else {
    if (lane == 0) {
      escore[widx] = part;
      atomicMax(&emax[d], fenc(part));
    }
  }
}

// ---------- pass 2: exp + denominator ----------
__global__ void edge_exp(float* __restrict__ escore, const unsigned* __restrict__ emax,
                         float* __restrict__ denom, const int* __restrict__ dstE,
                         int E_orig, int E_tot, int heads) {
  const int e = blockIdx.x * blockDim.x + threadIdx.x;
  if (e >= E_tot) return;
  const int d = (e < E_orig) ? dstE[e] : (e - E_orig);
  for (int h = 0; h < heads; ++h) {
    float ev = escore[(size_t)e * heads + h];
    float mx = fdec(emax[(size_t)d * heads + h]);
    float x  = __expf(ev - mx);
    escore[(size_t)e * heads + h] = x;
    atomicAdd(&denom[(size_t)d * heads + h], x);
  }
}

// ---------- pass 3: weighted aggregation (wave per edge) ----------
template <int HEADS>
__global__ void edge_aggregate(const float* __restrict__ xl, const float* __restrict__ escore,
                               const float* __restrict__ denom,
                               const int* __restrict__ srcE, const int* __restrict__ dstE,
                               float* __restrict__ agg, int E_orig, int E_tot) {
  const int widx = blockIdx.x * (blockDim.x >> 5) + (threadIdx.x >> 5);
  if (widx >= E_tot) return;
  const int lane = threadIdx.x & 31;
  int s, d;
  if (widx < E_orig) { s = srcE[widx]; d = dstE[widx]; } else { s = d = widx - E_orig; }

  const int h   = (HEADS == 2) ? (lane >> 4) : 0;      // k=4*lane: lanes 0-15 -> head 0
  const float ex = escore[(size_t)widx * HEADS + h];
  const float dn = denom[(size_t)d * HEADS + h];
  const float alpha = ex / (dn + 1e-16f);

  const float4 xv = *(const float4*)(xl + (size_t)s * 128 + lane * 4);
  float* out = agg + (size_t)d * 128 + lane * 4;
  atomicAdd(out + 0, xv.x * alpha);
  atomicAdd(out + 1, xv.y * alpha);
  atomicAdd(out + 2, xv.z * alpha);
  atomicAdd(out + 3, xv.w * alpha);
}

// ---------- layer-0 epilogue: +bias, LayerNorm, ELU (wave per node, in place) ----------
__global__ void ln_elu(float* __restrict__ h, const float* __restrict__ bias,
                       const float* __restrict__ gamma, const float* __restrict__ beta,
                       int n) {
  const int node = blockIdx.x * (blockDim.x >> 5) + (threadIdx.x >> 5);
  if (node >= n) return;
  const int lane = threadIdx.x & 31;
  float* row = h + (size_t)node * 128;

  float v[4];
  float s = 0.f;
  #pragma unroll
  for (int j = 0; j < 4; ++j) {
    int k = lane * 4 + j;
    v[j] = row[k] + bias[k];
    s += v[j];
  }
  s += __shfl_xor(s, 1); s += __shfl_xor(s, 2); s += __shfl_xor(s, 4);
  s += __shfl_xor(s, 8); s += __shfl_xor(s, 16);
  const float mu = s * (1.f / 128.f);

  float q = 0.f;
  #pragma unroll
  for (int j = 0; j < 4; ++j) { float t = v[j] - mu; q += t * t; }
  q += __shfl_xor(q, 1); q += __shfl_xor(q, 2); q += __shfl_xor(q, 4);
  q += __shfl_xor(q, 8); q += __shfl_xor(q, 16);
  const float rstd = rsqrtf(q * (1.f / 128.f) + 1e-5f);

  #pragma unroll
  for (int j = 0; j < 4; ++j) {
    int k = lane * 4 + j;
    float y = (v[j] - mu) * rstd * gamma[k] + beta[k];
    row[k] = y > 0.f ? y : (__expf(y) - 1.f);       // ELU
  }
}

// ---------- layer-1 epilogue: +bias, write node_emb, mean-pool atomics ----------
__global__ void bias_pool(float* __restrict__ node_emb, const float* __restrict__ bias,
                          const int* __restrict__ batch, float* __restrict__ sums,
                          float* __restrict__ cnt, int n) {
  const int node = blockIdx.x * (blockDim.x >> 5) + (threadIdx.x >> 5);
  if (node >= n) return;
  const int lane = threadIdx.x & 31;
  const int g = batch[node];
  float* row = node_emb + (size_t)node * 128;
  #pragma unroll
  for (int j = 0; j < 4; ++j) {
    int k = lane * 4 + j;
    float y = row[k] + bias[k];
    row[k] = y;
    atomicAdd(&sums[(size_t)g * 128 + k], y);
  }
  if (lane == 0) atomicAdd(&cnt[g], 1.0f);
}

__global__ void finalize_pool(const float* __restrict__ sums, const float* __restrict__ cnt,
                              float* __restrict__ graph_emb, int total) {
  const int i = blockIdx.x * blockDim.x + threadIdx.x;
  if (i < total) graph_emb[i] = sums[i] / fmaxf(cnt[i >> 7], 1.0f);
}

// ---------------------------------------------------------------------------
extern "C" void kernel_launch(void* const* d_in, const int* in_sizes, int n_in,
                              void* d_out, int out_size, void* d_ws, size_t ws_size,
                              hipStream_t stream) {
  const int N = in_sizes[0] / 128;
  const int E = in_sizes[1] / 2;
  const int G = 64;
  const int Etot = E + N;

  const float* x    = (const float*)d_in[0];
  const int*   src  = (const int*)d_in[1];
  const int*   dst  = src + E;
  const int*   batch = (const int*)d_in[2];
  const float* Wl0  = (const float*)d_in[4];
  const float* Wr0  = (const float*)d_in[5];
  const float* att0 = (const float*)d_in[6];
  const float* b0   = (const float*)d_in[7];
  const float* g0   = (const float*)d_in[8];
  const float* be0  = (const float*)d_in[9];
  const float* Wl1  = (const float*)d_in[10];
  const float* Wr1  = (const float*)d_in[11];
  const float* att1 = (const float*)d_in[12];
  const float* b1   = (const float*)d_in[13];

  float* node_out  = (float*)d_out;              // N*128
  float* graph_out = node_out + (size_t)N * 128; // G*128

  // workspace layout
  float* xl     = (float*)d_ws;                         // N*128
  float* xr     = xl + (size_t)N * 128;                 // N*128
  float* h      = xr + (size_t)N * 128;                 // N*128 (agg0 then h)
  float* escore = h + (size_t)N * 128;                  // Etot*2
  float* denom  = escore + (size_t)Etot * 2;            // N*2
  unsigned* emax = (unsigned*)(denom + (size_t)N * 2);  // N*2
  float* sums   = (float*)(emax + (size_t)N * 2);       // G*128
  float* cnt    = sums + (size_t)G * 128;               // G

  const int gemmBlocks = (N + 15) / 16;
  const int edgeWaves  = (Etot + 7) / 8;   // 8 waves / 256-thread block
  const int edgeThr    = (Etot + 255) / 256;
  const int nodeWaves  = (N + 7) / 8;

  // ---- init (inside capture; deterministic every call) ----
  hipMemsetAsync(h, 0, (size_t)N * 128 * sizeof(float), stream);
  hipMemsetAsync(denom, 0, (size_t)N * 2 * sizeof(float), stream);
  hipMemsetAsync(node_out, 0, (size_t)N * 128 * sizeof(float), stream);
  hipMemsetAsync(sums, 0, ((size_t)G * 128 + G) * sizeof(float), stream);
  init_u32<<<(N * 2 + 255) / 256, 256, 0, stream>>>(emax, ENC_NEG_INF, N * 2);

  // ---- layer 0: GATv2(128 -> 64, heads=2, concat) ----
  gemm128<<<gemmBlocks, 256, 0, stream>>>(x, Wl0, xl, N);
  gemm128<<<gemmBlocks, 256, 0, stream>>>(x, Wr0, xr, N);
  edge_score_max<2><<<edgeWaves, 256, 0, stream>>>(xl, xr, src, dst, att0, emax, escore, E, Etot);
  edge_exp<<<edgeThr, 256, 0, stream>>>(escore, emax, denom, dst, E, Etot, 2);
  edge_aggregate<2><<<edgeWaves, 256, 0, stream>>>(xl, escore, denom, src, dst, h, E, Etot);
  ln_elu<<<nodeWaves, 256, 0, stream>>>(h, b0, g0, be0, N);

  // ---- layer 1: GATv2(128 -> 128, heads=1) ----
  hipMemsetAsync(denom, 0, (size_t)N * sizeof(float), stream);
  init_u32<<<(N + 255) / 256, 256, 0, stream>>>(emax, ENC_NEG_INF, N);
  gemm128<<<gemmBlocks, 256, 0, stream>>>(h, Wl1, xl, N);
  gemm128<<<gemmBlocks, 256, 0, stream>>>(h, Wr1, xr, N);
  edge_score_max<1><<<edgeWaves, 256, 0, stream>>>(xl, xr, src, dst, att1, emax, escore, E, Etot);
  edge_exp<<<edgeThr, 256, 0, stream>>>(escore, emax, denom, dst, E, Etot, 1);
  edge_aggregate<1><<<edgeWaves, 256, 0, stream>>>(xl, escore, denom, src, dst, node_out, E, Etot);

  // ---- bias + global mean pool ----
  bias_pool<<<nodeWaves, 256, 0, stream>>>(node_out, b1, batch, sums, cnt, N);
  finalize_pool<<<(G * 128 + 255) / 256, 256, 0, stream>>>(sums, cnt, graph_out, G * 128);
}
